// TopoGradLoss_69054484185626
// MI455X (gfx1250) — matrix operand order repaired and checked by reading, not compiled, not run on hardware
//
#include <hip/hip_runtime.h>
#include <hip/hip_bf16.h>
#include <math.h>

// ---------------- problem constants (from the reference) ----------------
#define N_PTS   8192
#define DIM     32
#define K_KDE   32
#define K_RIPS  16
#define SCALE_F 64.0f
#define DESTNUM 10

typedef __attribute__((ext_vector_type(16))) _Float16 v16h;
typedef __attribute__((ext_vector_type(8)))  _Float16 v8h;
typedef __attribute__((ext_vector_type(8)))  float    v8f;
typedef __attribute__((ext_vector_type(4)))  float    v4f;
typedef __attribute__((ext_vector_type(4)))  unsigned int v4u;
typedef __attribute__((ext_vector_type(8)))  int      v8i;
typedef __attribute__((ext_vector_type(4)))  int      v4i;

#define NWAVES 4          // waves per block in the knn kernel
#define BLK    (NWAVES * 32)

// ---------------------------------------------------------------------------
// Kernel 0a: convert x to f16 once (saves per-tile cvt in the hot loop)
// ---------------------------------------------------------------------------
__global__ void tg_tof16(const float* __restrict__ x, _Float16* __restrict__ xh, int nelem) {
    int i = blockIdx.x * blockDim.x + threadIdx.x;
    if (i < nelem) xh[i] = (_Float16)x[i];
}

// ---------------------------------------------------------------------------
// Kernel 0b: row squared norms (fp32, exact)
// ---------------------------------------------------------------------------
__global__ void tg_sqnorm(const float* __restrict__ x, float* __restrict__ sqn, int n) {
    int i = blockIdx.x * blockDim.x + threadIdx.x;
    if (i < n) {
        const float* r = x + (size_t)i * DIM;
        float s = 0.f;
#pragma unroll
        for (int k = 0; k < DIM; ++k) s += r[k] * r[k];
        sqn[i] = s;
    }
}

// ---------------------------------------------------------------------------
// Kernel 1: fused WMMA distance tiles + per-point top-K_KDE selection.
// The block's 16 points form the loop-invariant B fragment (TDM-staged into
// LDS once); A sweeps all 512 row tiles, 2 tiles per iteration so two
// independent WMMA/select streams interleave and fill the WMMA->VALU hazard.
// C layout hands each lane its own point's candidates straight from the
// accumulator: lane l -> point i0 + (l&15), swept rows v + 8*(l>>4).
// ---------------------------------------------------------------------------
__global__ void __launch_bounds__(BLK)
tg_knn(const _Float16* __restrict__ xh, const float* __restrict__ sqn,
       float* __restrict__ kde_raw, int* __restrict__ rips) {
    __shared__ float    cd[K_KDE][BLK];       // per-lane candidate dists
    __shared__ int      ci[K_KDE][BLK];       // per-lane candidate indices
    __shared__ float    bestd[16][K_KDE];
    __shared__ int      besti[16][K_KDE];
    __shared__ _Float16 bstage[16 * DIM];     // TDM-staged B panel (16 pts x 32 f16)

    const int tid  = threadIdx.x;
    const int lane = tid & 31;
    const int wave = tid >> 5;
    const int i0   = blockIdx.x * 16;
    const int r16  = lane & 15, h16 = lane >> 4;

    // ---- Stage the block's B panel into LDS with the Tensor Data Mover.
    // D#: count=1, data_size=2B, 2-D tile 32x16, tensor row stride = 32 elems.
    if (wave == 0) {
        const unsigned long long ga =
            (unsigned long long)(size_t)(xh + (size_t)i0 * DIM);
        const unsigned lds_off = (unsigned)(size_t)(&bstage[0]);
        v4u g0;
        g0[0] = 1u;                                    // count = 1 (valid user D#)
        g0[1] = lds_off;                               // lds_addr
        g0[2] = (unsigned)(ga & 0xFFFFFFFFu);          // global_addr[31:0]
        g0[3] = (unsigned)((ga >> 32) & 0x01FFFFFFu)   // global_addr[56:32]
              | (2u << 30);                            // type = 2 ("image")
        v8i g1;
        g1[0] = (1 << 16);                 // wg_mask=0 | data_size=1 (2 bytes)
        g1[1] = (DIM << 16);               // tensor_dim0 = 32 (bits 48..63)
        g1[2] = ((N_PTS - i0) << 16);      // dim0 hi=0 | tensor_dim1 lo (bits 80..95)
        g1[3] = (DIM << 16);               // dim1 hi=0 | tile_dim0 = 32 (bits 112..127)
        g1[4] = 16;                        // tile_dim1 = 16 | tile_dim2 = 0
        g1[5] = DIM;                       // tensor_dim0_stride = 32 (lo32)
        g1[6] = 0;                         // stride hi | tensor_dim1_stride lo
        g1[7] = 0;
        v4i gz = {0, 0, 0, 0};
#if defined(__clang_major__) && __clang_major__ >= 23
        v8i gz8 = {0, 0, 0, 0, 0, 0, 0, 0};
        __builtin_amdgcn_tensor_load_to_lds(g0, g1, gz, gz, gz8, 0);
#else
        __builtin_amdgcn_tensor_load_to_lds(g0, g1, gz, gz, 0);
#endif
        __builtin_amdgcn_s_wait_tensorcnt(0);
    }
    __syncthreads();

    // ---- Loop-invariant B fragment (32x16 f16) from the staged panel:
    // lane<16: col N=lane, K=0..7,16..23 ; lane>=16: col N=lane-16, K=8..15,24..31
    const int kb = (lane < 16) ? 0 : 8;
    v16h b;
    {
        const v8h b0 = *(const v8h*)(bstage + (size_t)r16 * DIM + kb);
        const v8h b1 = *(const v8h*)(bstage + (size_t)r16 * DIM + kb + 16);
#pragma unroll
        for (int t = 0; t < 8; ++t) { b[t] = b0[t]; b[8 + t] = b1[t]; }
    }
    const float cs = sqn[i0 + r16];        // |point|^2 for this lane's point
    const int   kb0 = (lane < 16) ? 0 : 8; // A-fragment K runs

    // init per-lane selection lists
#pragma unroll
    for (int k = 0; k < K_KDE; ++k) { cd[k][tid] = 3.4e38f; ci[k][tid] = -1; }
    float curmax = 3.4e38f;

    // two tiles per iteration: jt and jt + NWAVES (stride 2*NWAVES)
    for (int jt = wave; jt < N_PTS / 16; jt += 2 * NWAVES) {
        const int j0 = jt * 16;
        const int j1 = (jt + NWAVES) * 16;
        {   // prefetch the next iteration's first A panel (global_prefetch_b8)
            int jn = j0 + 2 * NWAVES * 16; if (jn >= N_PTS) jn = 0;
            __builtin_prefetch(xh + (size_t)jn * DIM, 0, 1);
        }
        // ---- A fragments for both tiles (independent load streams)
        v16h a0v, a1v;
        {
            const _Float16* xr0 = xh + (size_t)(j0 + r16) * DIM;
            const _Float16* xr1 = xh + (size_t)(j1 + r16) * DIM;
            const v8h p0 = *(const v8h*)(xr0 + kb0);
            const v8h p1 = *(const v8h*)(xr0 + kb0 + 16);
            const v8h q0 = *(const v8h*)(xr1 + kb0);
            const v8h q1 = *(const v8h*)(xr1 + kb0 + 16);
#pragma unroll
            for (int t = 0; t < 8; ++t) {
                a0v[t] = p0[t]; a0v[8 + t] = p1[t];
                a1v[t] = q0[t]; a1v[8 + t] = q1[t];
            }
        }
        // swept-row squared norms for both tiles (contiguous 8 floats each)
        const v4f rA0 = *(const v4f*)(sqn + j0 + 8 * h16);
        const v4f rB0 = *(const v4f*)(sqn + j0 + 8 * h16 + 4);
        const v4f rA1 = *(const v4f*)(sqn + j1 + 8 * h16);
        const v4f rB1 = *(const v4f*)(sqn + j1 + 8 * h16 + 4);

        // ---- two back-to-back WMMAs (same B; no reuse hint: A/D differ)
        v8f c0 = {}, c1 = {};
        c0 = __builtin_amdgcn_wmma_f32_16x16x32_f16(
                 false, a0v, false, b, (short)0, c0, false, false);
        c1 = __builtin_amdgcn_wmma_f32_16x16x32_f16(
                 false, a1v, false, b, (short)0, c1, false, false);

        // ---- selection straight from the accumulators
#pragma unroll
        for (int v = 0; v < 8; ++v) {
            const float rn0 = (v < 4) ? rA0[v & 3] : rB0[v & 3];
            const float d0  = fmaxf(rn0 + cs - 2.0f * c0[v], 0.0f);
            const int   jj0 = j0 + 8 * h16 + v;
            if (d0 < curmax) {
                int mp = 0; float mv = -1.f;
#pragma unroll
                for (int k = 0; k < K_KDE; ++k) {
                    float v2 = cd[k][tid];
                    if (v2 > mv) { mv = v2; mp = k; }
                }
                cd[mp][tid] = d0; ci[mp][tid] = jj0;
                float nm = -1.f;
#pragma unroll
                for (int k = 0; k < K_KDE; ++k) nm = fmaxf(nm, cd[k][tid]);
                curmax = nm;
            }
        }
#pragma unroll
        for (int v = 0; v < 8; ++v) {
            const float rn1 = (v < 4) ? rA1[v & 3] : rB1[v & 3];
            const float d1  = fmaxf(rn1 + cs - 2.0f * c1[v], 0.0f);
            const int   jj1 = j1 + 8 * h16 + v;
            if (d1 < curmax) {
                int mp = 0; float mv = -1.f;
#pragma unroll
                for (int k = 0; k < K_KDE; ++k) {
                    float v2 = cd[k][tid];
                    if (v2 > mv) { mv = v2; mp = k; }
                }
                cd[mp][tid] = d1; ci[mp][tid] = jj1;
                float nm = -1.f;
#pragma unroll
                for (int k = 0; k < K_KDE; ++k) nm = fmaxf(nm, cd[k][tid]);
                curmax = nm;
            }
        }
    }
    __syncthreads();

    // ---- final merge: thread t (<16) owns point i0+t; sources are lanes
    // {w*32 + t, w*32 + 16 + t} for w = 0..NWAVES-1 (disjoint swept rows)
    if (tid < 16) {
#pragma unroll
        for (int k = 0; k < K_KDE; ++k) { bestd[tid][k] = 3.4e38f; besti[tid][k] = -1; }
        float bmax = 3.4e38f;
        for (int w = 0; w < NWAVES; ++w) {
            for (int half = 0; half < 2; ++half) {
                const int src = w * 32 + half * 16 + tid;
                for (int k = 0; k < K_KDE; ++k) {
                    const float d = cd[k][src];
                    if (d < bmax) {
                        int mp = 0; float mv = -1.f;
                        for (int q = 0; q < K_KDE; ++q) {
                            float v2 = bestd[tid][q];
                            if (v2 > mv) { mv = v2; mp = q; }
                        }
                        bestd[tid][mp] = d; besti[tid][mp] = ci[k][src];
                        float nm = -1.f;
                        for (int q = 0; q < K_KDE; ++q) nm = fmaxf(nm, bestd[tid][q]);
                        bmax = nm;
                    }
                }
            }
        }
        // sort the 32 survivors ascending (insertion sort)
        for (int a1 = 1; a1 < K_KDE; ++a1) {
            float dv = bestd[tid][a1]; int iv = besti[tid][a1];
            int b1 = a1 - 1;
            while (b1 >= 0 && bestd[tid][b1] > dv) {
                bestd[tid][b1 + 1] = bestd[tid][b1];
                besti[tid][b1 + 1] = besti[tid][b1];
                --b1;
            }
            bestd[tid][b1 + 1] = dv; besti[tid][b1 + 1] = iv;
        }
        // KDE = sum exp(-d2/scale) over K_KDE nearest (self included)
        float s = 0.f;
        for (int k = 0; k < K_KDE; ++k) s += __expf(-bestd[tid][k] * (1.0f / SCALE_F));
        kde_raw[i0 + tid] = s;
        for (int k = 0; k < K_RIPS; ++k)
            rips[(size_t)(i0 + tid) * K_RIPS + k] = besti[tid][k];
    }
}

// ---------------------------------------------------------------------------
// Kernel 2: max(kde_raw) and argmin(kde_raw) (argmin invariant under scaling)
// ---------------------------------------------------------------------------
__global__ void __launch_bounds__(1024)
tg_reduce(const float* __restrict__ kraw, float* __restrict__ maxv,
          int* __restrict__ argminI, int n) {
    __shared__ float smax[1024];
    __shared__ float smin[1024];
    __shared__ int   smini[1024];
    const int tid = threadIdx.x;
    float mx = -3.4e38f, mn = 3.4e38f; int mi = 0;
    for (int i = tid; i < n; i += 1024) {
        const float v = kraw[i];
        mx = fmaxf(mx, v);
        if (v < mn || (v == mn && i < mi)) { mn = v; mi = i; }
    }
    smax[tid] = mx; smin[tid] = mn; smini[tid] = mi;
    __syncthreads();
    for (int s = 512; s > 0; s >>= 1) {
        if (tid < s) {
            smax[tid] = fmaxf(smax[tid], smax[tid + s]);
            if (smin[tid + s] < smin[tid] ||
                (smin[tid + s] == smin[tid] && smini[tid + s] < smini[tid])) {
                smin[tid] = smin[tid + s]; smini[tid] = smini[tid + s];
            }
        }
        __syncthreads();
    }
    if (tid == 0) { maxv[0] = smax[0]; argminI[0] = smini[0]; }
}

// ---------------------------------------------------------------------------
// Kernel 3: normalize kde, seed sort arrays, init union-find state
// ---------------------------------------------------------------------------
__global__ void tg_norm_init(const float* __restrict__ kraw, const float* __restrict__ maxv,
                             float* __restrict__ kde, float* __restrict__ skey,
                             int* __restrict__ sidx, int* __restrict__ parent,
                             int* __restrict__ death, int n) {
    int i = blockIdx.x * blockDim.x + threadIdx.x;
    if (i < n) {
        const float v = kraw[i] / maxv[0];
        kde[i]  = v;
        skey[i] = v;
        sidx[i] = i;
        parent[i] = i;
        death[i]  = -1;
    }
}

// ---------------------------------------------------------------------------
// Kernel 4: one bitonic compare-exchange stage (descending sort by key)
// ---------------------------------------------------------------------------
__global__ void tg_bitonic(float* __restrict__ key, int* __restrict__ idx,
                           int j, int k, int n) {
    const unsigned i = blockIdx.x * blockDim.x + threadIdx.x;
    const unsigned ixj = i ^ (unsigned)j;
    if (i < (unsigned)n && ixj > i) {
        const float ki = key[i], kj = key[ixj];
        const bool blockUp = ((i & (unsigned)k) == 0);
        const bool doswap = blockUp ? (ki < kj) : (ki > kj);   // overall descending
        if (doswap) {
            key[i] = kj; key[ixj] = ki;
            int t = idx[i]; idx[i] = idx[ixj]; idx[ixj] = t;
        }
    }
}

__global__ void tg_rank(const int* __restrict__ order, int* __restrict__ rankA, int n) {
    int p = blockIdx.x * blockDim.x + threadIdx.x;
    if (p < n) rankA[order[p]] = p;
}

// ---------------------------------------------------------------------------
// Kernel 5: sequential ToMATo persistence scan — one wave; 16 lanes do the
// union-find root finds in parallel, shfl-argmax picks the surviving root.
// ---------------------------------------------------------------------------
__global__ void __launch_bounds__(32)
tg_persist(const float* __restrict__ kde, const int* __restrict__ order,
           const int* __restrict__ rankA, const int* __restrict__ rips,
           int* __restrict__ parent, int* __restrict__ death,
           const int* __restrict__ argminI, int n) {
    const int lane = threadIdx.x;
    for (int pos = 0; pos < n; ++pos) {
        const int i = order[pos];
        float score = -3.4e38f;
        int   gl = -1;
        if (lane < K_RIPS) {
            const int nb = rips[(size_t)i * K_RIPS + lane];
            if (rankA[nb] < pos) {
                int r = nb;
                int p = parent[r];
                while (p != r) {                 // path halving
                    const int gp = parent[p];
                    parent[r] = gp;
                    r = gp;
                    p = parent[r];
                }
                gl = r;
                score = kde[r];
            }
        }
        float bs = score; int bg = gl;
        for (int off = 16; off > 0; off >>= 1) {
            const float os = __shfl_xor(bs, off, 32);
            const int   og = __shfl_xor(bg, off, 32);
            if (os > bs || (os == bs && og > bg)) { bs = os; bg = og; }
        }
        const int g = (bg >= 0) ? bg : i;
        if (gl >= 0 && gl != g) {                // this neighbor's cluster dies
            death[gl] = i;
            parent[gl] = g;
        }
        if (lane == 0) parent[i] = g;
        __threadfence_block();
        __syncthreads();
    }
    if (lane == 0) death[order[0]] = argminI[0]; // essential pair
}

// ---------------------------------------------------------------------------
// Kernel 6: loss — top-DESTNUM persistence pairs via repeated argmax, then
// l_change + l_salient reduction.
// ---------------------------------------------------------------------------
__global__ void __launch_bounds__(256)
tg_loss(const float* __restrict__ kde, const int* __restrict__ death,
        float* __restrict__ out, int n) {
    __shared__ float sv[256];
    __shared__ int   si[256];
    __shared__ int   chosen[DESTNUM];
    const int tid = threadIdx.x;

    for (int t = 0; t < DESTNUM; ++t) {
        float bm = -3.4e38f; int bi = n;
        for (int i = tid; i < n; i += 256) {
            const int d = death[i];
            if (d < 0) continue;
            bool used = false;
            for (int u = 0; u < t; ++u) if (chosen[u] == i) used = true;
            if (used) continue;
            const float pers = kde[i] - kde[d];
            if (pers > bm || (pers == bm && i < bi)) { bm = pers; bi = i; }
        }
        sv[tid] = bm; si[tid] = bi;
        __syncthreads();
        for (int s = 128; s > 0; s >>= 1) {
            if (tid < s) {
                if (sv[tid + s] > sv[tid] ||
                    (sv[tid + s] == sv[tid] && si[tid + s] < si[tid])) {
                    sv[tid] = sv[tid + s]; si[tid] = si[tid + s];
                }
            }
            __syncthreads();
        }
        if (tid == 0) chosen[t] = si[0];
        __syncthreads();
    }

    float acc = 0.f;
    for (int i = tid; i < n; i += 256) {
        const int d = death[i];
        if (d < 0) continue;
        const float dv = kde[d];
        const float pers = kde[i] - dv;
        bool sal = false;
        for (int u = 0; u < DESTNUM; ++u) if (chosen[u] == i) sal = true;
        acc += sal ? ((kde[i] - 1.f) * (kde[i] - 1.f) + dv * dv) : pers * pers;
    }
    sv[tid] = acc;
    __syncthreads();
    for (int s = 128; s > 0; s >>= 1) {
        if (tid < s) sv[tid] += sv[tid + s];
        __syncthreads();
    }
    if (tid == 0) out[0] = sv[0];
}

// ---------------------------------------------------------------------------
extern "C" void kernel_launch(void* const* d_in, const int* in_sizes, int n_in,
                              void* d_out, int out_size, void* d_ws, size_t ws_size,
                              hipStream_t stream) {
    (void)in_sizes; (void)n_in; (void)out_size; (void)ws_size;
    const int n = N_PTS;
    const float* x = (const float*)d_in[0];
    float* out = (float*)d_out;

    // workspace layout (~1.3 MB)
    _Float16* xh   = (_Float16*)d_ws;              // N*DIM f16 (512 KB)
    float* sqn     = (float*)(xh + (size_t)n * DIM);   // N
    float* kde_raw = sqn + n;                      // N
    float* kde     = kde_raw + n;                  // N
    float* skey    = kde + n;                      // N
    int*   sidx    = (int*)(skey + n);             // N (becomes `order` after sort)
    int*   rips    = sidx + n;                     // N * K_RIPS
    int*   rankA   = rips + (size_t)n * K_RIPS;    // N
    int*   parent  = rankA + n;                    // N
    int*   death   = parent + n;                   // N
    float* maxv    = (float*)(death + n);          // 1
    int*   argminI = (int*)(maxv + 1);             // 1

    tg_tof16<<<(n * DIM) / 256, 256, 0, stream>>>(x, xh, n * DIM);
    tg_sqnorm<<<n / 256, 256, 0, stream>>>(x, sqn, n);
    tg_knn<<<n / 16, BLK, 0, stream>>>(xh, sqn, kde_raw, rips);
    tg_reduce<<<1, 1024, 0, stream>>>(kde_raw, maxv, argminI, n);
    tg_norm_init<<<n / 256, 256, 0, stream>>>(kde_raw, maxv, kde, skey, sidx,
                                              parent, death, n);
    for (int k = 2; k <= n; k <<= 1)
        for (int j = k >> 1; j > 0; j >>= 1)
            tg_bitonic<<<n / 256, 256, 0, stream>>>(skey, sidx, j, k, n);
    tg_rank<<<n / 256, 256, 0, stream>>>(sidx, rankA, n);
    tg_persist<<<1, 32, 0, stream>>>(kde, sidx, rankA, rips, parent, death, argminI, n);
    tg_loss<<<1, 256, 0, stream>>>(kde, death, out, n);
}
